// MoELayer_35605278884297
// MI455X (gfx1250) — compile-verified, matrix operand order checked by reading, not run on hardware
//
#include <hip/hip_runtime.h>
#include <math.h>

// ---------------- problem constants (match reference) ----------------
#define DDIM 1024
#define EEXP 8
#define HDIM 4096
#define TILE_M 32             // two 16-row WMMA m-tiles per block
#define HCHUNK 512
#define XSTRIDE (DDIM + 8)    // 1032 u16 -> 2064B row stride, 4-dword bank skew
#define HSTRIDE (HCHUNK + 8)  // 520 u16  -> 1040B row stride, 4-dword bank skew

typedef __bf16 bf16_t;
typedef bf16_t v16bf __attribute__((ext_vector_type(16)));
typedef float  v8f   __attribute__((ext_vector_type(8)));
typedef unsigned short u16;

// round float to bf16 (nearest, ties-away): bias the mantissa
__device__ __forceinline__ unsigned bfr(float f) {
    union { float f; unsigned u; } a; a.f = f;
    return a.u + 0x8000u;
}
// pack two floats into one dword of 2 bf16 via v_perm_b32 (3 VALU total)
__device__ __forceinline__ unsigned pkbf(float lo, float hi) {
    return __builtin_amdgcn_perm(bfr(hi), bfr(lo), 0x07060302u);
}

union FragU { v16bf v; uint4 q[2]; unsigned w[8]; };

// A-fragment (16x32 bf16, M x K) from row-major bf16(u16) LDS tile.
// ISA 7.12.2: lane<16 -> M=lane, K=[0..7]+[16..23]; lane>=16 -> K=[8..15]+[24..31].
__device__ __forceinline__ v16bf frag_from_lds(const u16* base, int stride,
                                               int mbase, int kbase, int lane) {
    const int row  = mbase + (lane & 15);
    const int koff = kbase + ((lane >> 4) << 3);
    const u16* p = base + row * stride + koff;
    FragU u;
    u.q[0] = *(const uint4*)(p);        // ds_load_b128
    u.q[1] = *(const uint4*)(p + 16);   // ds_load_b128
    return u.v;
}

// B-fragment (32x16 bf16, K x N) = W^T, W row-major [Nout, K].
// fp32 source: pack pairs with v_perm_b32 on the fly.
__device__ __forceinline__ v16bf frag_from_w(const float* W, int ldw,
                                             int nbase, int kbase, int lane) {
    const float* p = W + (size_t)(nbase + (lane & 15)) * (size_t)ldw
                       + (size_t)(kbase + ((lane >> 4) << 3));
    float4 f0 = *(const float4*)(p);
    float4 f1 = *(const float4*)(p + 4);
    float4 f2 = *(const float4*)(p + 16);
    float4 f3 = *(const float4*)(p + 20);
    FragU u;
    u.w[0] = pkbf(f0.x, f0.y);  u.w[1] = pkbf(f0.z, f0.w);
    u.w[2] = pkbf(f1.x, f1.y);  u.w[3] = pkbf(f1.z, f1.w);
    u.w[4] = pkbf(f2.x, f2.y);  u.w[5] = pkbf(f2.z, f2.w);
    u.w[6] = pkbf(f3.x, f3.y);  u.w[7] = pkbf(f3.z, f3.w);
    return u.v;
}
// bf16 source (pre-converted weights): two global_load_b128, zero VALU.
__device__ __forceinline__ v16bf frag_from_w(const u16* W, int ldw,
                                             int nbase, int kbase, int lane) {
    const u16* p = W + (size_t)(nbase + (lane & 15)) * (size_t)ldw
                     + (size_t)(kbase + ((lane >> 4) << 3));
    FragU u;
    u.q[0] = *(const uint4*)(p);
    u.q[1] = *(const uint4*)(p + 16);
    return u.v;
}

// staging helpers: produce one packed bf16x2 word of the x tile
__device__ __forceinline__ unsigned stage_word(const float* xr, int colw) {
    const float2 f = *(const float2*)(xr + 2 * colw);
    return pkbf(f.x, f.y);
}
__device__ __forceinline__ unsigned stage_word(const u16* xr, int colw) {
    return *(const unsigned*)(xr + 2 * colw);
}

// branchless tanh-form GELU: 0.5*v*(1+tanh(0.79788456*(v+0.044715 v^3)))
__device__ __forceinline__ float gelu_f(float v) {
    const float u  = 0.7978845608028654f * v * __builtin_fmaf(0.044715f * v, v, 1.0f);
    const float e  = __expf(2.0f * u);          // v_exp_f32; inf/0 extremes OK
    const float th = 1.0f - 2.0f / (e + 1.0f);  // tanh(u), branch-free
    return 0.5f * v * (1.0f + th);
}

// ---------------- kernel 0: bulk fp32 -> packed bf16 conversion ----------------
__global__ __launch_bounds__(256) void f32_to_bf16(const float* __restrict__ src,
                                                   unsigned* __restrict__ dst,
                                                   int nwords) {
    const int stride = gridDim.x * 256;
    for (int i = blockIdx.x * 256 + threadIdx.x; i < nwords; i += stride) {
        const float2 f = *(const float2*)(src + 2 * (size_t)i);
        dst[i] = pkbf(f.x, f.y);
    }
}

// ---------------- kernel 1: router (top-2 + softmax weights + expert lists) ----
__global__ __launch_bounds__(256) void moe_router(
        const float* __restrict__ x, const float* __restrict__ gate_w,
        const float* __restrict__ gate_b, int* __restrict__ counts,
        int* __restrict__ tlist, float* __restrict__ wlist, int N)
{
    __shared__ float gw[EEXP * DDIM];   // 32 KB
    const int tid = threadIdx.x;
    for (int i = tid; i < EEXP * DDIM; i += 256) gw[i] = gate_w[i];
    __syncthreads();

    const int t = blockIdx.x * 256 + tid;
    if (t >= N) return;

    const float* xr = x + (size_t)t * DDIM;
    float acc[EEXP];
#pragma unroll
    for (int e = 0; e < EEXP; ++e) acc[e] = gate_b[e];
    for (int d = 0; d < DDIM; ++d) {
        const float xv = xr[d];
#pragma unroll
        for (int e = 0; e < EEXP; ++e) acc[e] += xv * gw[e * DDIM + d];
    }
    // top-2, first occurrence wins on ties (jax.lax.top_k semantics)
    int i0 = 0; float v0 = acc[0];
#pragma unroll
    for (int e = 1; e < EEXP; ++e) if (acc[e] > v0) { v0 = acc[e]; i0 = e; }
    int i1 = (i0 == 0) ? 1 : 0; float v1 = acc[i1];
#pragma unroll
    for (int e = 0; e < EEXP; ++e)
        if (e != i0 && acc[e] > v1) { v1 = acc[e]; i1 = e; }

    const float w0 = 1.0f / (1.0f + __expf(v1 - v0));   // softmax over {v0, v1}
    const float w1 = 1.0f - w0;

    int p0 = atomicAdd(&counts[i0], 1);
    tlist[i0 * N + p0] = t;  wlist[i0 * N + p0] = w0;
    int p1 = atomicAdd(&counts[i1], 1);
    tlist[i1 * N + p1] = t;  wlist[i1 * N + p1] = w1;
}

// ---------------- kernel 2: load-balancing loss ----------------
__global__ __launch_bounds__(256) void moe_lbloss(
        const float* __restrict__ gate_w, const int* __restrict__ counts,
        float* __restrict__ loss_out, int N)
{
    __shared__ float red[EEXP][257];
    const int tid = threadIdx.x;
    float acc[EEXP];
#pragma unroll
    for (int e = 0; e < EEXP; ++e) acc[e] = 0.0f;

    for (int d = tid; d < DDIM; d += 256) {
        float g[EEXP]; float m = -INFINITY;
#pragma unroll
        for (int e = 0; e < EEXP; ++e) { g[e] = gate_w[e * DDIM + d]; m = fmaxf(m, g[e]); }
        float s = 0.0f;
#pragma unroll
        for (int e = 0; e < EEXP; ++e) { g[e] = __expf(g[e] - m); s += g[e]; }
        const float inv = 1.0f / s;
#pragma unroll
        for (int e = 0; e < EEXP; ++e) acc[e] += g[e] * inv;
    }
#pragma unroll
    for (int e = 0; e < EEXP; ++e) red[e][tid] = acc[e];
    __syncthreads();
    for (int off = 128; off > 0; off >>= 1) {
        if (tid < off)
#pragma unroll
            for (int e = 0; e < EEXP; ++e) red[e][tid] += red[e][tid + off];
        __syncthreads();
    }
    if (tid == 0) {
        float loss = 0.0f;
#pragma unroll
        for (int e = 0; e < EEXP; ++e)
            loss += ((float)counts[e] / (float)N) * (red[e][0] / (float)DDIM);
        loss_out[0] = (float)EEXP * loss;
    }
}

// ---------------- kernel 3: fused expert FFN via bf16 WMMA ----------------
// Block = 256 threads = 8 wave32s; one (expert, 32-token) tile per block.
// XT/WT = u16 (pre-converted bf16, pure-load path) or float (convert on the fly).
template <typename XT, typename WT>
__global__ __launch_bounds__(256) void moe_expert(
        const XT* __restrict__ x,
        const WT* __restrict__ w1, const float* __restrict__ b1,
        const WT* __restrict__ w2, const float* __restrict__ b2,
        const int* __restrict__ counts, const int* __restrict__ tlist,
        const float* __restrict__ wlist, float* __restrict__ out, int N)
{
    __shared__ __align__(16) u16 lds_x[TILE_M * XSTRIDE];  // 64.5 KB
    __shared__ __align__(16) u16 lds_h[TILE_M * HSTRIDE];  // 32.5 KB
    __shared__ int   s_tok[TILE_M];
    __shared__ float s_w[TILE_M];

    const int e     = blockIdx.y;
    const int cnt   = counts[e];
    const int tile0 = blockIdx.x * TILE_M;
    if (tile0 >= cnt) return;                    // block-uniform exit

    const int tid  = threadIdx.x;
    const int lane = tid & 31;
    const int wave = tid >> 5;                   // 0..7

    if (tid < TILE_M) {
        const int idx   = tile0 + tid;
        const int valid = idx < cnt;
        const int src   = e * N + (valid ? idx : tile0);  // pad rows clamp, weight 0
        s_tok[tid] = tlist[src];
        s_w[tid]   = valid ? wlist[src] : 0.0f;
    }
    __syncthreads();

    // stage x tile (32 x 1024) into packed-bf16 LDS, coalesced
    for (int i = tid; i < TILE_M * (DDIM / 2); i += 256) {
        const int row  = i >> 9;             // / (DDIM/2)
        const int colw = i & (DDIM / 2 - 1);
        *(unsigned*)(&lds_x[row * XSTRIDE + 2 * colw]) =
            stage_word(x + (size_t)s_tok[row] * DDIM, colw);
    }
    __syncthreads();

    const WT* w1e = w1 + (size_t)e * HDIM * DDIM;
    const WT* w2e = w2 + (size_t)e * DDIM * HDIM;
    const float* b1e = b1 + (size_t)e * HDIM;
    const float* b2e = b2 + (size_t)e * DDIM;

    // y accumulators: wave owns D columns [wave*128, +128) -> 8 n-tiles x 2 m-tiles
    v8f yacc0[8], yacc1[8];
#pragma unroll
    for (int t = 0; t < 8; ++t) {
        const float bv = b2e[wave * 128 + t * 16 + (lane & 15)];
#pragma unroll
        for (int r = 0; r < 8; ++r) { yacc0[t][r] = bv; yacc1[t][r] = bv; }
    }

    for (int c = 0; c < HDIM / HCHUNK; ++c) {
        // ---- GEMM1: h_chunk = gelu(x @ w1[c*512 .. +512]^T + b1); wave owns 64 cols
        v8f acc0[4], acc1[4];
#pragma unroll
        for (int t = 0; t < 4; ++t) {
            const float bv = b1e[c * HCHUNK + wave * 64 + t * 16 + (lane & 15)];
#pragma unroll
            for (int r = 0; r < 8; ++r) { acc0[t][r] = bv; acc1[t][r] = bv; }
        }
        for (int k = 0; k < DDIM; k += 32) {
            const v16bf a0 = frag_from_lds(lds_x, XSTRIDE, 0,  k, lane);
            const v16bf a1 = frag_from_lds(lds_x, XSTRIDE, 16, k, lane);
#pragma unroll
            for (int t = 0; t < 4; ++t) {
                const v16bf b = frag_from_w(w1e, DDIM,
                                            c * HCHUNK + wave * 64 + t * 16, k, lane);
                acc0[t] = __builtin_amdgcn_wmma_f32_16x16x32_bf16(
                              false, a0, false, b, (short)0, acc0[t], false, false);
                acc1[t] = __builtin_amdgcn_wmma_f32_16x16x32_bf16(
                              false, a1, false, b, (short)0, acc1[t], false, false);
            }
        }
        {
            const int row_off = (lane >> 4) << 3;
#pragma unroll
            for (int t = 0; t < 4; ++t) {
                const int col = wave * 64 + t * 16 + (lane & 15);
#pragma unroll
                for (int r = 0; r < 8; ++r) {
                    lds_h[(r + row_off) * HSTRIDE + col] =
                        (u16)(bfr(gelu_f(acc0[t][r])) >> 16);
                    lds_h[(16 + r + row_off) * HSTRIDE + col] =
                        (u16)(bfr(gelu_f(acc1[t][r])) >> 16);
                }
            }
        }
        __syncthreads();

        // ---- GEMM2 partial: y += h_chunk @ w2[:, c*512 .. +512]^T
        for (int k = 0; k < HCHUNK; k += 32) {
            const v16bf a0 = frag_from_lds(lds_h, HSTRIDE, 0,  k, lane);
            const v16bf a1 = frag_from_lds(lds_h, HSTRIDE, 16, k, lane);
#pragma unroll
            for (int t = 0; t < 8; ++t) {
                const v16bf b = frag_from_w(w2e, HDIM,
                                            wave * 128 + t * 16, c * HCHUNK + k, lane);
                yacc0[t] = __builtin_amdgcn_wmma_f32_16x16x32_bf16(
                               false, a0, false, b, (short)0, yacc0[t], false, false);
                yacc1[t] = __builtin_amdgcn_wmma_f32_16x16x32_bf16(
                               false, a1, false, b, (short)0, yacc1[t], false, false);
            }
        }
        __syncthreads();   // lds_h rewritten next chunk
    }

    // epilogue: out[tok, :] += y * routing_weight (pad rows weight 0)
    const int row_off = (lane >> 4) << 3;
#pragma unroll
    for (int t = 0; t < 8; ++t) {
        const int col = wave * 128 + t * 16 + (lane & 15);
#pragma unroll
        for (int r = 0; r < 8; ++r) {
            const int row = r + row_off;
            atomicAdd(&out[(size_t)s_tok[row] * DDIM + col], yacc0[t][r] * s_w[row]);
            atomicAdd(&out[(size_t)s_tok[16 + row] * DDIM + col],
                      yacc1[t][r] * s_w[16 + row]);
        }
    }
}

// ---------------- launcher ----------------
extern "C" void kernel_launch(void* const* d_in, const int* in_sizes, int n_in,
                              void* d_out, int out_size, void* d_ws, size_t ws_size,
                              hipStream_t stream) {
    const float* x      = (const float*)d_in[0];
    const float* gate_w = (const float*)d_in[1];
    const float* gate_b = (const float*)d_in[2];
    const float* w1     = (const float*)d_in[3];
    const float* b1     = (const float*)d_in[4];
    const float* w2     = (const float*)d_in[5];
    const float* b2     = (const float*)d_in[6];
    float* out = (float*)d_out;

    const int N = in_sizes[0] / DDIM;   // 8192 tokens

    // workspace layout:
    //   [counts: E ints][tlist: E*N ints][wlist: E*N floats]  (~512 KB)
    //   [bw1: E*H*D u16][bw2: E*D*H u16][bx: N*D u16]          (~151 MB, optional)
    int*   counts = (int*)d_ws;
    int*   tlist  = counts + EEXP;
    float* wlist  = (float*)(tlist + (size_t)EEXP * N);

    const size_t lists_bytes =
        sizeof(int) * EEXP + (sizeof(int) + sizeof(float)) * (size_t)EEXP * N;
    const size_t bf_off = (lists_bytes + 255) & ~(size_t)255;
    const size_t n_w = (size_t)EEXP * HDIM * DDIM;     // elems per weight tensor
    const size_t n_x = (size_t)N * DDIM;
    const size_t need = bf_off + 2 * (2 * n_w + n_x);  // bf16 bytes

    hipMemsetAsync(counts, 0, EEXP * sizeof(int), stream);
    hipMemsetAsync(d_out, 0, (size_t)out_size * sizeof(float), stream);

    moe_router<<<dim3((N + 255) / 256), dim3(256), 0, stream>>>(
        x, gate_w, gate_b, counts, tlist, wlist, N);

    moe_lbloss<<<dim3(1), dim3(256), 0, stream>>>(
        gate_w, counts, out + (size_t)N * DDIM, N);

    if (ws_size >= need) {
        // fast path: one-shot bf16 conversion of weights + activations (~18 us of
        // HBM traffic), then a pure-load WMMA pipeline (no per-tile conversion VALU)
        u16* bw1 = (u16*)((char*)d_ws + bf_off);
        u16* bw2 = bw1 + n_w;
        u16* bx  = bw2 + n_w;
        f32_to_bf16<<<dim3(4096), dim3(256), 0, stream>>>(w1, (unsigned*)bw1,
                                                          (int)(n_w / 2));
        f32_to_bf16<<<dim3(4096), dim3(256), 0, stream>>>(w2, (unsigned*)bw2,
                                                          (int)(n_w / 2));
        f32_to_bf16<<<dim3(1024), dim3(256), 0, stream>>>(x, (unsigned*)bx,
                                                          (int)(n_x / 2));
        moe_expert<u16, u16><<<dim3(N / TILE_M, EEXP), dim3(256), 0, stream>>>(
            bx, bw1, b1, bw2, b2, counts, tlist, wlist, out, N);
    } else {
        // fallback: convert fp32 weights to bf16 in-register per fragment
        moe_expert<float, float><<<dim3(N / TILE_M, EEXP), dim3(256), 0, stream>>>(
            x, w1, b1, w2, b2, counts, tlist, wlist, out, N);
    }
}
